// BahdanauAttention_42365557408326
// MI455X (gfx1250) — matrix operand order repaired and checked by reading, not compiled, not run on hardware
//
#include <hip/hip_runtime.h>
#include <hip/hip_bf16.h>

// Problem constants (match reference setup_inputs)
#define Hdim 1024
#define Tdim 2048
#define Bdim 32

typedef __attribute__((ext_vector_type(8)))  __bf16 v8bf;
typedef __attribute__((ext_vector_type(16))) __bf16 v16bf;
typedef __attribute__((ext_vector_type(8)))  float  v8f;

// Branch-free tanh on hardware transcendentals (v_exp_f32 + v_rcp_f32).
// tanh(x) = 1 - 2*rcp(exp(2x)+1)
// exp(2x)->inf => rcp(inf)=0 => 1.0 ; exp(2x)->0 => rcp(1)=1 => -1.0.
// __builtin_amdgcn_rcpf emits a single v_rcp_f32 (no IEEE div expansion).
__device__ __forceinline__ float fast_tanh(float x) {
    float e = __expf(2.0f * x);
    return 1.0f - 2.0f * __builtin_amdgcn_rcpf(e + 1.0f);
}

// ---------------------------------------------------------------------------
// Kernel 1: convert Wa_w (H x H fp32, row-major over h) to bf16 scratch.
// 2 MB result lives in L2 (192 MB) for the whole scores GEMM.
// ---------------------------------------------------------------------------
__global__ void wconv_kernel(const float* __restrict__ w, __bf16* __restrict__ wb) {
    int idx = blockIdx.x * 256 + threadIdx.x;            // H*H / 4 threads
    const float4 f = *(const float4*)(w + (size_t)idx * 4);
    __bf16* o = wb + (size_t)idx * 4;
    o[0] = (__bf16)f.x; o[1] = (__bf16)f.y; o[2] = (__bf16)f.z; o[3] = (__bf16)f.w;
}

// ---------------------------------------------------------------------------
// Kernel 2: dproj2[b][o] = dec[b]·Ua_w[o] + Ua_b[o] + Wa_b[o]   (tiny GEMV)
// ---------------------------------------------------------------------------
__global__ void dproj_kernel(const float* __restrict__ dec,
                             const float* __restrict__ Ua_w,
                             const float* __restrict__ Ua_b,
                             const float* __restrict__ Wa_b,
                             float* __restrict__ dproj2) {
    int idx = blockIdx.x * 256 + threadIdx.x;            // B*H threads
    int b = idx >> 10, o = idx & (Hdim - 1);
    const float4* dr = (const float4*)(dec + (size_t)b * Hdim);
    const float4* ur = (const float4*)(Ua_w + (size_t)o * Hdim);
    float acc = 0.f;
    #pragma unroll 4
    for (int k = 0; k < Hdim / 4; ++k) {
        float4 d = dr[k], u = ur[k];
        acc += d.x * u.x + d.y * u.y + d.z * u.z + d.w * u.w;
    }
    dproj2[idx] = acc + Ua_b[o] + Wa_b[o];
}

// ---------------------------------------------------------------------------
// Kernel 3: WMMA score kernel.
// Block = 256 threads = 8 wave32s. Each block owns a 16-row t-tile of one
// batch. X tile staged in LDS as bf16; W streamed from bf16 L2 copy.
// scores[b][t] = v_b + sum_o v_w[o] * tanh( X[t]·W[o] + dproj2[b][o] )
// ---------------------------------------------------------------------------
__global__ void __launch_bounds__(256)
scores_kernel(const float* __restrict__ enc,
              const __bf16* __restrict__ Wb,
              const float* __restrict__ dproj2,
              const float* __restrict__ v_w,
              const float* __restrict__ v_b,
              float* __restrict__ scores) {
    __shared__ __bf16 Xs[16][1032];     // 16 t-rows x H, +8 pad (33 KB)
    __shared__ float  scW[8][16];       // per-wave partial scores

    const int b    = blockIdx.x;        // batch
    const int t0   = blockIdx.y * 16;   // t-tile base
    const int tid  = threadIdx.x;
    const int lane = tid & 31;
    const int wn   = tid >> 5;          // wave id = n-tile 0..7
    const int half = lane >> 4;         // 0: lanes 0-15, 1: lanes 16-31
    const int nn   = lane & 15;

    // ---- Phase 1: load + convert X tile (coalesced float4) ----
    const float* Xg = enc + ((size_t)b * Tdim + t0) * Hdim;
    for (int i = tid * 4; i < 16 * Hdim; i += 256 * 4) {
        int row = i >> 10, k = i & (Hdim - 1);
        const float4 f = *(const float4*)(Xg + (size_t)row * Hdim + k);
        Xs[row][k]     = (__bf16)f.x;
        Xs[row][k + 1] = (__bf16)f.y;
        Xs[row][k + 2] = (__bf16)f.z;
        Xs[row][k + 3] = (__bf16)f.w;
    }
    __syncthreads();

    // A fragment source row for this lane (ISA 16-bit A layout: M = lane&15)
    const __bf16* arow = &Xs[nn][0];

    float sacc[8];
    #pragma unroll
    for (int r = 0; r < 8; ++r) sacc[r] = 0.f;

    // ---- Phase 2: 8 o-tiles x (K=1024 in steps of 32) of bf16 WMMA ----
    for (int obase = 0; obase < Hdim; obase += 128) {
        const int o = obase + wn * 16 + nn;              // this lane's column
        const __bf16* wrow = Wb + (size_t)o * Hdim;      // contiguous over k

        v8f c = {};
        #pragma unroll 4
        for (int k = 0; k < Hdim; k += 32) {
            // A: 16x32 bf16, lane holds K = half*8+{0..7} and 16+half*8+{0..7}
            v8bf alo = *(const v8bf*)(arow + k + half * 8);
            v8bf ahi = *(const v8bf*)(arow + k + 16 + half * 8);
            v16bf a = __builtin_shufflevector(alo, ahi,
                0, 1, 2, 3, 4, 5, 6, 7, 8, 9, 10, 11, 12, 13, 14, 15);
            // B: 32x16 bf16, lane holds column n, K = half*8+{0..7} and 16+half*8+{0..7}
            v8bf blo = *(const v8bf*)(wrow + k + half * 8);
            v8bf bhi = *(const v8bf*)(wrow + k + 16 + half * 8);
            v16bf bb = __builtin_shufflevector(blo, bhi,
                0, 1, 2, 3, 4, 5, 6, 7, 8, 9, 10, 11, 12, 13, 14, 15);
            c = __builtin_amdgcn_wmma_f32_16x16x32_bf16(
                    false, a, false, bb, (short)0, c, false, false);
        }

        // Fused epilogue: branch-free tanh + dot with v_w, per-lane accum.
        const float dpv = dproj2[(size_t)b * Hdim + o];
        const float vv  = v_w[o];
        #pragma unroll
        for (int r = 0; r < 8; ++r)                      // C: m = r + 8*half
            sacc[r] += vv * fast_tanh(c[r] + dpv);
    }

    // ---- Phase 3: reduce over o. Lanes sharing the same m are the 16 lanes
    // of one half-wave -> shfl_xor tree, then cross-wave sum via LDS. ----
    #pragma unroll
    for (int r = 0; r < 8; ++r) {
        float v = sacc[r];
        v += __shfl_xor(v, 8);
        v += __shfl_xor(v, 4);
        v += __shfl_xor(v, 2);
        v += __shfl_xor(v, 1);
        if (nn == 0) scW[wn][half * 8 + r] = v;          // t-local = r + 8*half
    }
    __syncthreads();

    if (tid < 16) {
        float s = v_b[0];
        #pragma unroll
        for (int w8 = 0; w8 < 8; ++w8) s += scW[w8][tid];
        scores[(size_t)b * Tdim + t0 + tid] = s;
    }
}

// ---------------------------------------------------------------------------
// Kernel 4: softmax over T per batch (fixed-order tree reductions).
// ---------------------------------------------------------------------------
__global__ void softmax_kernel(const float* __restrict__ scores,
                               float* __restrict__ attn) {
    __shared__ float red[256];
    const int b = blockIdx.x, tid = threadIdx.x;
    const float* s = scores + (size_t)b * Tdim;

    float lmax = -INFINITY;
    for (int t = tid; t < Tdim; t += 256) lmax = fmaxf(lmax, s[t]);
    red[tid] = lmax; __syncthreads();
    for (int k = 128; k > 0; k >>= 1) {
        if (tid < k) red[tid] = fmaxf(red[tid], red[tid + k]);
        __syncthreads();
    }
    const float m = red[0];
    __syncthreads();

    float lsum = 0.f;
    for (int t = tid; t < Tdim; t += 256) lsum += __expf(s[t] - m);
    red[tid] = lsum; __syncthreads();
    for (int k = 128; k > 0; k >>= 1) {
        if (tid < k) red[tid] += red[tid + k];
        __syncthreads();
    }
    const float inv = 1.0f / red[0];

    float* a = attn + (size_t)b * Tdim;
    for (int t = tid; t < Tdim; t += 256) a[t] = __expf(s[t] - m) * inv;
}

// ---------------------------------------------------------------------------
// Kernel 5: context partials. Split T into 8 chunks for parallelism;
// fixed-order accumulation (no float atomics -> deterministic).
// ---------------------------------------------------------------------------
__global__ void ctx_partial_kernel(const float* __restrict__ attn,
                                   const float* __restrict__ enc,
                                   float* __restrict__ part) {
    const int bid = blockIdx.x;                  // 32 * 8 * 4 = 1024 blocks
    const int b  = bid >> 5;
    const int tc = (bid & 31) >> 2;              // 0..7 t-chunk
    const int hc = bid & 3;                      // 0..3 h-chunk
    const int h  = hc * 256 + threadIdx.x;

    const float* aw = attn + (size_t)b * Tdim + tc * 256;
    const float* ew = enc + ((size_t)b * Tdim + (size_t)tc * 256) * Hdim + h;
    float acc = 0.f;
    #pragma unroll 8
    for (int tt = 0; tt < 256; ++tt)
        acc += aw[tt] * ew[(size_t)tt * Hdim];
    part[((size_t)b * 8 + tc) * Hdim + h] = acc;
}

__global__ void ctx_reduce_kernel(const float* __restrict__ part,
                                  float* __restrict__ ctx) {
    int idx = blockIdx.x * 256 + threadIdx.x;    // B*H
    int b = idx >> 10, h = idx & (Hdim - 1);
    float s = 0.f;
    #pragma unroll
    for (int tc = 0; tc < 8; ++tc)
        s += part[((size_t)b * 8 + tc) * Hdim + h];
    ctx[idx] = s;
}

// ---------------------------------------------------------------------------
extern "C" void kernel_launch(void* const* d_in, const int* in_sizes, int n_in,
                              void* d_out, int out_size, void* d_ws, size_t ws_size,
                              hipStream_t stream) {
    const float* dec   = (const float*)d_in[0];   // [B,H]
    const float* enc   = (const float*)d_in[1];   // [B,T,H]
    const float* Wa_w  = (const float*)d_in[2];   // [H,H]
    const float* Wa_b  = (const float*)d_in[3];   // [H]
    const float* Ua_w  = (const float*)d_in[4];   // [H,H]
    const float* Ua_b  = (const float*)d_in[5];   // [H]
    const float* v_w   = (const float*)d_in[6];   // [H]
    const float* v_b   = (const float*)d_in[7];   // [1]

    float* ctx_out  = (float*)d_out;                       // [B,H]
    float* attn_out = (float*)d_out + Bdim * Hdim;         // [B,T]

    // Scratch layout (bytes): dproj2 | W_bf16 | scores | ctx partials
    char*  ws     = (char*)d_ws;
    float* dproj2 = (float*)(ws);                           // 128 KB
    __bf16* wb16  = (__bf16*)(ws + 131072);                 // 2 MB (32B aligned)
    float* scores = (float*)(ws + 2228224);                 // 256 KB
    float* part   = (float*)(ws + 2490368);                 // 1 MB

    wconv_kernel<<<Hdim * Hdim / 1024, 256, 0, stream>>>(Wa_w, wb16);
    dproj_kernel<<<Bdim * Hdim / 256, 256, 0, stream>>>(dec, Ua_w, Ua_b, Wa_b, dproj2);
    scores_kernel<<<dim3(Bdim, Tdim / 16), 256, 0, stream>>>(enc, wb16, dproj2,
                                                             v_w, v_b, scores);
    softmax_kernel<<<Bdim, 256, 0, stream>>>(scores, attn_out);
    ctx_partial_kernel<<<Bdim * 8 * 4, 256, 0, stream>>>(attn_out, enc, part);
    ctx_reduce_kernel<<<Bdim * Hdim / 256, 256, 0, stream>>>(part, ctx_out);
}
